// QueryInitialization_31903017074872
// MI455X (gfx1250) — matrix operand order repaired and checked by reading, not compiled, not run on hardware
//
#include <hip/hip_runtime.h>
#include <hip/hip_bf16.h>
#include <stdint.h>

typedef _Float16 f16;
typedef __attribute__((ext_vector_type(16))) _Float16 v16h;
typedef __attribute__((ext_vector_type(8)))  _Float16 v8h;
typedef __attribute__((ext_vector_type(8)))  float    v8f;

#define BATCH   16
#define CDIM    256
#define HDIM    100
#define NPIX    10000            // 100*100
#define POSHW   50
#define NDETQ   100
#define NRECQ   25

// ---------------------------------------------------------------------------
// Bilinear coordinate helper (align_corners=False, out=100, in=50 -> scale 0.5)
// ---------------------------------------------------------------------------
__device__ __forceinline__ void bicoord(int o, int nin, float scale,
                                        int& i0, int& i1, float& t) {
    float src = ((float)o + 0.5f) * scale - 0.5f;
    src = fminf(fmaxf(src, 0.0f), (float)(nin - 1));
    i0 = (int)floorf(src);
    i1 = min(i0 + 1, nin - 1);
    t  = src - (float)i0;
}

__device__ __forceinline__ float pos_bilerp(const float* __restrict__ pe, int c,
                                            int y0, int y1, float ty,
                                            int x0, int x1, float tx) {
    const float* p = pe + c * (POSHW * POSHW);
    float p00 = p[y0 * POSHW + x0];
    float p01 = p[y0 * POSHW + x1];
    float p10 = p[y1 * POSHW + x0];
    float p11 = p[y1 * POSHW + x1];
    float r0 = p00 + (p01 - p00) * tx;
    float r1 = p10 + (p11 - p10) * tx;
    return r0 + (r1 - r0) * ty;
}

// ---------------------------------------------------------------------------
// Kernel 1: streaming projection. One thread per pixel; reads x once,
// builds feat = x + bilinear(pos_embed), emits cls_logits, bbox_pred, conf.
// ---------------------------------------------------------------------------
__global__ void qi_proj_kernel(const float* __restrict__ x,
                               const float* __restrict__ pos_embed,
                               const float* __restrict__ Wc, const float* __restrict__ bc,
                               const float* __restrict__ Wb, const float* __restrict__ bb,
                               float* __restrict__ cls_out,
                               float* __restrict__ bbox_out,
                               float* __restrict__ conf_out) {
    __shared__ float sW[6 * CDIM];   // rows: Wc0, Wc1, Wb0..Wb3
    __shared__ float sb[6];
    for (int i = threadIdx.x; i < 2 * CDIM; i += blockDim.x) sW[i] = Wc[i];
    for (int i = threadIdx.x; i < 4 * CDIM; i += blockDim.x) sW[2 * CDIM + i] = Wb[i];
    if (threadIdx.x < 2) sb[threadIdx.x] = bc[threadIdx.x];
    if (threadIdx.x < 4) sb[2 + threadIdx.x] = bb[threadIdx.x];
    __syncthreads();

    int pix = blockIdx.x * blockDim.x + threadIdx.x;
    if (pix >= BATCH * NPIX) return;
    int b = pix / NPIX;
    int n = pix - b * NPIX;
    int py = n / HDIM, px = n - py * HDIM;

    int y0, y1, x0, x1; float ty, tx;
    bicoord(py, POSHW, 0.5f, y0, y1, ty);
    bicoord(px, POSHW, 0.5f, x0, x1, tx);

    const float* xb = x + (size_t)b * CDIM * NPIX + n;
    float acc0 = 0.f, acc1 = 0.f, acc2 = 0.f, acc3 = 0.f, acc4 = 0.f, acc5 = 0.f;
    #pragma unroll 4
    for (int c = 0; c < CDIM; ++c) {
        float f = xb[(size_t)c * NPIX] + pos_bilerp(pos_embed, c, y0, y1, ty, x0, x1, tx);
        acc0 = fmaf(f, sW[c], acc0);
        acc1 = fmaf(f, sW[CDIM + c], acc1);
        acc2 = fmaf(f, sW[2 * CDIM + c], acc2);
        acc3 = fmaf(f, sW[3 * CDIM + c], acc3);
        acc4 = fmaf(f, sW[4 * CDIM + c], acc4);
        acc5 = fmaf(f, sW[5 * CDIM + c], acc5);
    }
    float l0 = acc0 + sb[0], l1 = acc1 + sb[1];
    cls_out[(size_t)pix * 2 + 0] = l0;
    cls_out[(size_t)pix * 2 + 1] = l1;
    bbox_out[(size_t)pix * 4 + 0] = acc2 + sb[2];
    bbox_out[(size_t)pix * 4 + 1] = acc3 + sb[3];
    bbox_out[(size_t)pix * 4 + 2] = acc4 + sb[4];
    bbox_out[(size_t)pix * 4 + 3] = acc5 + sb[5];
    // softmax(...)[:,1] == sigmoid(l1 - l0), numerically stable form
    float m = fmaxf(l0, l1);
    float e0 = __expf(l0 - m), e1 = __expf(l1 - m);
    conf_out[pix] = e1 / (e0 + e1);
}

// ---------------------------------------------------------------------------
// Kernel 2: per-batch top-100 (sorted desc, smallest index on ties — matches
// jax.lax.top_k). conf copy lives in LDS (40 KB, fine in 320 KB WGP LDS).
// top-25 is a prefix of this list.
// ---------------------------------------------------------------------------
__global__ void qi_topk_kernel(const float* __restrict__ conf, int* __restrict__ idx_out) {
    __shared__ float sconf[NPIX];
    __shared__ float rv[1024];
    __shared__ int   ri[1024];
    int b = blockIdx.x;
    int tid = threadIdx.x;
    for (int i = tid; i < NPIX; i += blockDim.x) sconf[i] = conf[(size_t)b * NPIX + i];
    __syncthreads();

    for (int k = 0; k < NDETQ; ++k) {
        float best = -3.0e38f;
        int bi = NPIX;
        for (int i = tid; i < NPIX; i += blockDim.x) {
            float v = sconf[i];
            if (v > best || (v == best && i < bi)) { best = v; bi = i; }
        }
        rv[tid] = best; ri[tid] = bi;
        __syncthreads();
        for (int s = 512; s > 0; s >>= 1) {
            if (tid < s) {
                float ov = rv[tid + s]; int oi = ri[tid + s];
                if (ov > rv[tid] || (ov == rv[tid] && oi < ri[tid])) { rv[tid] = ov; ri[tid] = oi; }
            }
            __syncthreads();
        }
        if (tid == 0) {
            idx_out[b * NDETQ + k] = ri[0];
            sconf[ri[0]] = -3.0e38f;
        }
        __syncthreads();
    }
}

// ---------------------------------------------------------------------------
// Kernel 3: gather selected rows by recomputing feat (x + pos) for the 1600
// selected pixels; emit f16 activations for the WMMA GEMMs.
// ---------------------------------------------------------------------------
__global__ void qi_gather_kernel(const float* __restrict__ x,
                                 const float* __restrict__ pos_embed,
                                 const int* __restrict__ idx,
                                 f16* __restrict__ Xdet, f16* __restrict__ Xrec) {
    int r = blockIdx.x;              // 0..1599
    int b = r / NDETQ, rank = r - b * NDETQ;
    int n = idx[r];
    int py = n / HDIM, px = n - py * HDIM;
    int y0, y1, x0, x1; float ty, tx;
    bicoord(py, POSHW, 0.5f, y0, y1, ty);
    bicoord(px, POSHW, 0.5f, x0, x1, tx);
    int c = threadIdx.x;             // 0..255
    float f = x[(size_t)b * CDIM * NPIX + (size_t)c * NPIX + n]
            + pos_bilerp(pos_embed, c, y0, y1, ty, x0, x1, tx);
    f16 h = (f16)f;
    Xdet[(size_t)r * CDIM + c] = h;
    if (rank < NRECQ) Xrec[((size_t)b * NRECQ + rank) * CDIM + c] = h;
}

// f32 -> f16 weight repack
__global__ void qi_cvt_f16(const float* __restrict__ src, f16* __restrict__ dst, int n) {
    int i = blockIdx.x * blockDim.x + threadIdx.x;
    if (i < n) dst[i] = (f16)src[i];
}

// ---------------------------------------------------------------------------
// Kernel 4: Y = act(X @ W^T + bias) [+ query], via v_wmma_f32_16x16x32_f16.
// X: [M,256] f16 row-major. W: [256,256] f16 row-major (out-major), so the
// WMMA B-matrix element B[k][n] = W[n][k] is a contiguous K-run of W row n.
// Wave32 fragment layouts per CDNA5 ISA §7.12.2:
//   A 16x32:  lane m=L&15, half=L>>4; elems 0..7 -> K=half*8+e, 8..15 -> K=16+half*8+e
//   B 32x16:  lane n=L&15, half=L>>4; elems 0..7 -> K=half*8+e, 8..15 -> K=16+half*8+e
//   C 16x16:  lane n=L&15; VGPR v -> M = v + 8*(L>>4)
// Block = 256 threads = 8 waves; each wave owns one 16x16 output tile.
// ---------------------------------------------------------------------------
__global__ void qi_mlp_wmma(const f16* __restrict__ X, const f16* __restrict__ W,
                            const float* __restrict__ bias,
                            const float* __restrict__ query, int qrows,
                            f16* __restrict__ Yh, float* __restrict__ Yf,
                            int relu) {
    const int wave  = threadIdx.x >> 5;
    const int lane  = threadIdx.x & 31;
    const int mtile = blockIdx.x;
    const int ntile = blockIdx.y * 8 + wave;
    const int half  = lane >> 4;
    const int mn    = lane & 15;

    const f16* Xrow = X + ((size_t)mtile * 16 + mn) * CDIM;   // A row m
    const f16* Wrow = W + ((size_t)ntile * 16 + mn) * CDIM;   // B column n = W row n

    union AV { v16h v; v8h h[2]; };
    v8f acc = {};
    #pragma unroll
    for (int kk = 0; kk < 8; ++kk) {
        const int kb = kk * 32;
        AV a, bmat;
        a.h[0]    = *(const v8h*)(Xrow + kb + half * 8);        // K = kb + half*8 .. +7
        a.h[1]    = *(const v8h*)(Xrow + kb + 16 + half * 8);   // K = kb+16+half*8 .. +7
        bmat.h[0] = *(const v8h*)(Wrow + kb + half * 8);        // K = kb + half*8 .. +7
        bmat.h[1] = *(const v8h*)(Wrow + kb + 16 + half * 8);   // K = kb+16+half*8 .. +7
        acc = __builtin_amdgcn_wmma_f32_16x16x32_f16(
            /*neg_a=*/false, a.v, /*neg_b=*/false, bmat.v,
            /*c_mod=*/(short)0, acc, /*reuse_a=*/false, /*reuse_b=*/false);
    }

    const int n  = ntile * 16 + mn;
    const float bn = bias[n];
    #pragma unroll
    for (int v = 0; v < 8; ++v) {
        int m = mtile * 16 + v + half * 8;
        float val = acc[v] + bn;
        if (relu) val = fmaxf(val, 0.0f);
        if (Yh) Yh[(size_t)m * CDIM + n] = (f16)val;
        if (Yf) {
            float q = query ? query[(size_t)(m % qrows) * CDIM + n] : 0.0f;
            Yf[(size_t)m * CDIM + n] = val + q;
        }
    }
}

// ---------------------------------------------------------------------------
extern "C" void kernel_launch(void* const* d_in, const int* in_sizes, int n_in,
                              void* d_out, int out_size, void* d_ws, size_t ws_size,
                              hipStream_t stream) {
    const float* x    = (const float*)d_in[0];
    const float* Wc   = (const float*)d_in[1];
    const float* bc   = (const float*)d_in[2];
    const float* Wb   = (const float*)d_in[3];
    const float* bb   = (const float*)d_in[4];
    const float* dW1  = (const float*)d_in[5];
    const float* db1  = (const float*)d_in[6];
    const float* dW2  = (const float*)d_in[7];
    const float* db2  = (const float*)d_in[8];
    const float* dW3  = (const float*)d_in[9];
    const float* db3  = (const float*)d_in[10];
    const float* rW1  = (const float*)d_in[11];
    const float* rb1  = (const float*)d_in[12];
    const float* rW2  = (const float*)d_in[13];
    const float* rb2  = (const float*)d_in[14];
    const float* rW3  = (const float*)d_in[15];
    const float* rb3  = (const float*)d_in[16];
    const float* detq = (const float*)d_in[17];
    const float* recq = (const float*)d_in[18];
    const float* pos  = (const float*)d_in[19];

    float* out   = (float*)d_out;
    float* det_o = out;                                   // 16*100*256
    float* rec_o = out + BATCH * NDETQ * CDIM;            // 16*25*256
    float* cls_o = rec_o + BATCH * NRECQ * CDIM;          // 16*10000*2
    float* bbx_o = cls_o + BATCH * NPIX * 2;              // 16*10000*4

    // workspace layout (256B aligned slabs)
    uintptr_t w = (uintptr_t)d_ws;
    auto alloc = [&](size_t bytes) -> uintptr_t {
        uintptr_t p = w; w += (bytes + 255) & ~(size_t)255; return p;
    };
    float* conf = (float*)alloc(BATCH * NPIX * sizeof(float));
    int*   idx  = (int*)  alloc(BATCH * NDETQ * sizeof(int));
    f16* Xdet = (f16*)alloc((size_t)BATCH * NDETQ * CDIM * sizeof(f16));
    f16* H1d  = (f16*)alloc((size_t)BATCH * NDETQ * CDIM * sizeof(f16));
    f16* H2d  = (f16*)alloc((size_t)BATCH * NDETQ * CDIM * sizeof(f16));
    f16* Xrec = (f16*)alloc((size_t)BATCH * NRECQ * CDIM * sizeof(f16));
    f16* H1r  = (f16*)alloc((size_t)BATCH * NRECQ * CDIM * sizeof(f16));
    f16* H2r  = (f16*)alloc((size_t)BATCH * NRECQ * CDIM * sizeof(f16));
    const int WN = CDIM * CDIM;
    f16* hdW1 = (f16*)alloc(WN * sizeof(f16));
    f16* hdW2 = (f16*)alloc(WN * sizeof(f16));
    f16* hdW3 = (f16*)alloc(WN * sizeof(f16));
    f16* hrW1 = (f16*)alloc(WN * sizeof(f16));
    f16* hrW2 = (f16*)alloc(WN * sizeof(f16));
    f16* hrW3 = (f16*)alloc(WN * sizeof(f16));
    (void)ws_size; (void)in_sizes; (void)n_in; (void)out_size;

    // weight repack (tiny, overlaps nothing of consequence)
    dim3 cb(256), cg((WN + 255) / 256);
    qi_cvt_f16<<<cg, cb, 0, stream>>>(dW1, hdW1, WN);
    qi_cvt_f16<<<cg, cb, 0, stream>>>(dW2, hdW2, WN);
    qi_cvt_f16<<<cg, cb, 0, stream>>>(dW3, hdW3, WN);
    qi_cvt_f16<<<cg, cb, 0, stream>>>(rW1, hrW1, WN);
    qi_cvt_f16<<<cg, cb, 0, stream>>>(rW2, hrW2, WN);
    qi_cvt_f16<<<cg, cb, 0, stream>>>(rW3, hrW3, WN);

    // streaming projection: 160000 pixels
    qi_proj_kernel<<<(BATCH * NPIX + 255) / 256, 256, 0, stream>>>(
        x, pos, Wc, bc, Wb, bb, cls_o, bbx_o, conf);

    // per-batch top-100
    qi_topk_kernel<<<BATCH, 1024, 0, stream>>>(conf, idx);

    // gather selected rows (recompute feat for 1600 pixels)
    qi_gather_kernel<<<BATCH * NDETQ, CDIM, 0, stream>>>(x, pos, idx, Xdet, Xrec);

    // det MLP: M=1600 -> grid (100, 2); rec MLP: M=400 -> grid (25, 2)
    dim3 mb(256);
    dim3 gd(NDETQ * BATCH / 16, 2);
    qi_mlp_wmma<<<gd, mb, 0, stream>>>(Xdet, hdW1, db1, nullptr, 1, H1d, nullptr, 1);
    qi_mlp_wmma<<<gd, mb, 0, stream>>>(H1d,  hdW2, db2, nullptr, 1, H2d, nullptr, 1);
    qi_mlp_wmma<<<gd, mb, 0, stream>>>(H2d,  hdW3, db3, detq, NDETQ, nullptr, det_o, 0);

    dim3 gr(NRECQ * BATCH / 16, 2);
    qi_mlp_wmma<<<gr, mb, 0, stream>>>(Xrec, hrW1, rb1, nullptr, 1, H1r, nullptr, 1);
    qi_mlp_wmma<<<gr, mb, 0, stream>>>(H1r,  hrW2, rb2, nullptr, 1, H2r, nullptr, 1);
    qi_mlp_wmma<<<gr, mb, 0, stream>>>(H2r,  hrW3, rb3, recq, NRECQ, nullptr, rec_o, 0);
}
